// DynamicRNN_39719857553996
// MI455X (gfx1250) — matrix operand order repaired and checked by reading, not compiled, not run on hardware
//
#include <hip/hip_runtime.h>

typedef __attribute__((ext_vector_type(16))) __bf16 bf16x16;
typedef __attribute__((ext_vector_type(8)))  __bf16 bf16x8;
typedef __attribute__((ext_vector_type(8)))  float  f32x8;

#define HDIM   256
#define DDIM   256
#define KTOT   512            // D + H
#define NKT    16             // KTOT / 32
#define FOURH  1024
#define KP     264            // 256 + 8 bf16 pad -> 528B row stride (16B aligned, 4-bank skew)

__device__ __forceinline__ float sigm(float v) { return 1.0f / (1.0f + __expf(-v)); }

// One-time repack of W (fp32 [512,1024] row-major) into bf16 WMMA-B fragment layout:
// frag(nt, kt): lane L holds column N = nt*16 + (L&15), elements e=0..15 are
// K = kt*32 + (L>>4)*16 + e.  Flat index = ((nt*NKT + kt)*32 + L)*16 + e.
__global__ void pack_w_kernel(const float* __restrict__ W, __bf16* __restrict__ Wp) {
    int tid  = blockIdx.x * blockDim.x + threadIdx.x;     // 0 .. 524287
    int e    = tid & 15;
    int lane = (tid >> 4) & 31;
    int kt   = (tid >> 9) & 15;
    int nt   = tid >> 13;
    int K = kt * 32 + ((lane >> 4) << 4) + e;
    int N = nt * 16 + (lane & 15);
    Wp[tid] = (__bf16)W[(size_t)K * FOURH + N];
}

// Persistent LSTM: 1 workgroup per 16 batch rows, 16 waves, each wave owns one
// 16-wide hidden tile and its i/j/f/o gate tiles.  c and h state in VGPRs.
__global__ __launch_bounds__(512) void lstm_persistent_kernel(
    const float* __restrict__ x,        // [B, T, D]
    const float* __restrict__ bias,     // [4H]
    const int*   __restrict__ seq_len,  // [B]
    const __bf16* __restrict__ Wp,      // packed W fragments (1 MB)
    float* __restrict__ out,            // [B, T, H]
    int T)
{
    __shared__ __bf16 xb[16 * KP];        // bf16 x_t tile, K = 0..255
    __shared__ __bf16 hb[2][16 * KP];     // double-buffered bf16 h tile, K = 0..255

    const int tid  = threadIdx.x;
    const int lane = tid & 31;
    const int ht   = tid >> 5;            // wave id == hidden tile 0..15
    const int b0   = blockIdx.x * 16;
    const int hi   = lane >> 4;           // half-wave: 0 or 1
    const int nlo  = lane & 15;

    // zero h buffer 0 (initial hidden state)
    for (int i = tid; i < 16 * KP; i += 512) hb[0][i] = (__bf16)0.0f;

    // per-lane constants (bias per gate column, seq_len per owned row)
    float bias_g[4];
#pragma unroll
    for (int q = 0; q < 4; ++q) bias_g[q] = bias[q * HDIM + ht * 16 + nlo];
    int sl[8];
#pragma unroll
    for (int g = 0; g < 8; ++g) sl[g] = seq_len[b0 + g + hi * 8];

    // Single per-lane base pointer for this wave's W fragments; fragment (q, kt)
    // lives at compile-time byte offset q*256KB + kt*1KB (+0 / +16 for the halves).
    const __bf16* wflat = Wp + (size_t)ht * NKT * 512 + lane * 16;

    f32x8 c    = {};   // cell state tile (C/D layout: row M = g + hi*8, col = nlo)
    f32x8 hrow = {};   // hidden state tile (fp32 copy for masking)

    __syncthreads();

    for (int t = 0; t < T; ++t) {
        // Opaque zero offset: keeps the 128 B-fragment global loads inside the
        // t-loop (address unknown per iteration -> no LICM -> no spills) while
        // preserving the pointer's global-addrspace provenance (global_load,
        // LOADcnt only, instead of flat_load which also ties up DScnt).
        int wofs = 0;
        asm volatile("" : "+v"(wofs));
        const __bf16* wbase = wflat + wofs;

        // ---- stage x_t tile into LDS as bf16 (coalesced, 8 elems/thread) ----
#pragma unroll
        for (int i = 0; i < 8; ++i) {
            int idx = tid + i * 512;
            int r = idx >> 8, k = idx & 255;
            xb[r * KP + k] = (__bf16)x[((size_t)(b0 + r) * T + t) * DDIM + k];
        }
        __syncthreads();

        const __bf16* hcur = hb[t & 1];

        f32x8 acc[4] = {{}, {}, {}, {}};   // zero C -> inline-0 SRC2 on first WMMA

        // ---- gates = [x_t, h] @ W : 16 K-tiles x 4 gate tiles of WMMA ----
#pragma unroll
        for (int kt = 0; kt < NKT; ++kt) {
            const __bf16* src = (kt < 8)
                ? &xb[nlo * KP + kt * 32 + hi * 16]
                : &hcur[nlo * KP + (kt - 8) * 32 + hi * 16];
            // A layout matches packed B: lane holds K kb..kb+15 contiguous (kb = kt*32 + hi*16)
            bf16x8 alo = *(const bf16x8*)(src);
            bf16x8 ahi = *(const bf16x8*)(src + 8);
            bf16x16 a = __builtin_shufflevector(alo, ahi,
                0,1,2,3,4,5,6,7,8,9,10,11,12,13,14,15);
#pragma unroll
            for (int q = 0; q < 4; ++q) {
                bf16x16 bfrag = *(const bf16x16*)(wbase + ((size_t)q * 16 * NKT + kt) * 512);
                acc[q] = __builtin_amdgcn_wmma_f32_16x16x32_bf16(
                    false, a, false, bfrag, (short)0, acc[q], false, false);
            }
        }

        // ---- elementwise LSTM cell + seq_len masking; update h in LDS ----
        __bf16* hnext = hb[(t + 1) & 1];
#pragma unroll
        for (int g = 0; g < 8; ++g) {
            float iv = acc[0][g] + bias_g[0];
            float jv = acc[1][g] + bias_g[1];
            float fv = acc[2][g] + bias_g[2];
            float ov = acc[3][g] + bias_g[3];
            float nc = c[g] * sigm(fv + 1.0f) + sigm(iv) * tanhf(jv);
            float nh = tanhf(nc) * sigm(ov);
            bool valid = t < sl[g];
            c[g]    = valid ? nc : c[g];
            hrow[g] = valid ? nh : hrow[g];
            hnext[(g + hi * 8) * KP + ht * 16 + nlo] = (__bf16)hrow[g];
            out[((size_t)(b0 + g + hi * 8) * T + t) * HDIM + ht * 16 + nlo] =
                valid ? nh : 0.0f;
        }
        __syncthreads();   // protect xb refill + hnext visibility for next step
    }
}

extern "C" void kernel_launch(void* const* d_in, const int* in_sizes, int n_in,
                              void* d_out, int out_size, void* d_ws, size_t ws_size,
                              hipStream_t stream) {
    const float* x       = (const float*)d_in[0];   // [B,T,D]
    const float* W       = (const float*)d_in[1];   // [D+H, 4H]
    const float* b       = (const float*)d_in[2];   // [4H]
    const int*   seq_len = (const int*)d_in[3];     // [B]
    float*       out     = (float*)d_out;           // [B,T,H]

    const int B = in_sizes[3];
    const int T = in_sizes[0] / (B * DDIM);

    __bf16* Wp = (__bf16*)d_ws;                     // needs 512*1024*2 = 1 MB

    // Phase 0: repack W into WMMA-B fragment layout (bf16).
    pack_w_kernel<<<(KTOT * FOURH) / 256, 256, 0, stream>>>(W, Wp);

    // Phase 1: persistent batch-parallel LSTM (B/16 workgroups, 16 waves each).
    lstm_persistent_kernel<<<B / 16, 512, 0, stream>>>(x, b, seq_len, Wp, out, T);
}
